// Model_Plus_79250736546082
// MI455X (gfx1250) — matrix operand order repaired and checked by reading, not compiled, not run on hardware
//
#include <hip/hip_runtime.h>

// ---------------------------------------------------------------------------
// Problem constants (from reference)
// ---------------------------------------------------------------------------
static constexpr int N_NODES = 50000;
static constexpr int F_IN    = 1024;
static constexpr int HID     = 256;
static constexpr int OUT     = 280;
static constexpr int OUTP    = 288;   // OUT padded to multiple of 32
static constexpr int S_SP    = 5000;
static constexpr int NNZ     = 800000;

typedef __attribute__((ext_vector_type(16))) __bf16 v16bf;
typedef __attribute__((ext_vector_type(8)))  float  v8f;

union FragBF16 {
    v16bf v;
    uint4 q[2];
    unsigned short u[16];
};

// ---------------------------------------------------------------------------
// Helpers
// ---------------------------------------------------------------------------
__device__ __forceinline__ unsigned short f2bf(float f) {
    unsigned int u = __float_as_uint(f);
    unsigned int r = u + 0x7FFFu + ((u >> 16) & 1u);   // round-to-nearest-even
    return (unsigned short)(r >> 16);
}

__global__ __launch_bounds__(256) void k_zero(float* p, long n) {
    long i = (long)blockIdx.x * 256 + threadIdx.x;
    long stride = (long)gridDim.x * 256;
    for (; i < n; i += stride) p[i] = 0.0f;
}

__global__ __launch_bounds__(256) void k_f32_to_bf16(const float* __restrict__ src,
                                                     unsigned short* __restrict__ dst, long n) {
    long i = (long)blockIdx.x * 256 + threadIdx.x;
    long stride = (long)gridDim.x * 256;
    for (; i < n; i += stride) dst[i] = f2bf(src[i]);
}

// Transpose+pad: w (K x M, f32, row-major) -> dst (Mpad x K, bf16)
// dst[m*K + k] = bf16(w[k*M + m]), zero for m >= M.
__global__ __launch_bounds__(256) void k_pad_weight_t(const float* __restrict__ w,
                                                      unsigned short* __restrict__ dst,
                                                      int K, int M, int Mpad) {
    long n = (long)Mpad * K;
    long i = (long)blockIdx.x * 256 + threadIdx.x;
    long stride = (long)gridDim.x * 256;
    for (; i < n; i += stride) {
        int m = (int)(i / K), k = (int)(i % K);
        dst[i] = (m < M) ? f2bf(w[(long)k * M + m]) : (unsigned short)0;
    }
}

// ---------------------------------------------------------------------------
// WMMA bf16 GEMM:  C[nrows x M] = act(A[nrows x K] @ B[K x M] + bias)
// B supplied pre-TRANSPOSED and padded: BT[Mpad x K] bf16.
// Block: 256 threads = 8 waves; tile 128(rows) x 32(cols); K-step 64
// (two 16x16x32 wmma sub-steps per stage, 2 col-tiles per wave).
// Double-buffered LDS, 1 barrier per K-step; staging split into a pure
// global-load phase (5 b128 loads in flight) then an LDS-store phase.
// Fragment reads are aligned ds_load_b128 (contiguous K runs);
// LDS row stride 72 (x2B = 144B) => 16B aligned, conflict-free banks.
// C layout per CDNA5 ISA 7.12.2; A/B frags use contiguous K (k = 16*lh + s) in both.
// ---------------------------------------------------------------------------
template<int ACT>
__global__ __launch_bounds__(256) void k_gemm_bf16(
    const unsigned short* __restrict__ A, const unsigned short* __restrict__ BT,
    const float* __restrict__ bias, float* __restrict__ C,
    int nrows, int K, int M, int ldOut, float slope)
{
    __shared__ unsigned short As[2][128 * 72];   // 128 rows x 64 k (stride 72)
    __shared__ unsigned short Bs[2][32 * 72];    // 32 cols(n) x 64 k (stride 72)

    const int tid  = threadIdx.x;
    const int wave = tid >> 5;
    const int lane = tid & 31;
    const int lh   = lane >> 4;     // lane half (0/1)
    const int l15  = lane & 15;
    const int rowBase = blockIdx.y * 128;
    const int colBase = blockIdx.x * 32;

    const int kb  = lh * 16;        // A-frag K base within sub-step
    const int klo = lh * 16;        // B-frag K base within sub-step

    const bool fullRows = (rowBase + 128 <= nrows);

    // per-thread staging coordinates (it-th A chunk: row arow[it], col-group acg)
    const int acg  = tid & 7;              // A k-group (8 bf16 each)
    const int arw  = tid >> 3;             // A row for it=0 (rows step by 32 per it)
    const int bn   = tid >> 3;             // B column (n) index
    const int bcg  = tid & 7;              // B k-group

    v8f acc0 = {}; v8f acc1 = {};

    auto stage = [&](int buf, int k0) {
        uint4 va[4];
        uint4 vb;
        // ---- phase 1: issue all global loads (stay in flight together)
        if (fullRows) {
            #pragma unroll
            for (int it = 0; it < 4; ++it) {
                int r = arw + it * 32;
                va[it] = *(const uint4*)(A + (long)(rowBase + r) * K + k0 + acg * 8);
            }
        } else {
            #pragma unroll
            for (int it = 0; it < 4; ++it) {
                int r = arw + it * 32;
                int grow = rowBase + r;
                va[it] = make_uint4(0u, 0u, 0u, 0u);
                if (grow < nrows)
                    va[it] = *(const uint4*)(A + (long)grow * K + k0 + acg * 8);
            }
        }
        vb = *(const uint4*)(BT + (long)(colBase + bn) * K + k0 + bcg * 8);

        // ---- phase 2: LDS stores
        unsigned short* as = &As[buf][0];
        unsigned short* bs = &Bs[buf][0];
        #pragma unroll
        for (int it = 0; it < 4; ++it) {
            int r = arw + it * 32;
            *(uint4*)(&as[r * 72 + acg * 8]) = va[it];
        }
        *(uint4*)(&bs[bn * 72 + bcg * 8]) = vb;
    };

    stage(0, 0);
    int cur = 0;

    for (int k0 = 0; k0 < K; k0 += 64) {
        __syncthreads();                    // buffer 'cur' ready for compute
        if (k0 + 64 < K) {
            stage(cur ^ 1, k0 + 64);        // overlap: fill other buffer
            if (fullRows && k0 + 128 < K) { // uniform-guarded prefetch, 2 tiles ahead
                #pragma unroll
                for (int it = 0; it < 4; ++it) {
                    const unsigned short* gp =
                        A + (long)(rowBase + arw + it * 32) * K + k0 + 128 + acg * 8;
                    __builtin_prefetch(gp, 0, 1);   // global_prefetch_b8
                }
            }
        }

        const unsigned short* asr = &As[cur][(wave * 16 + l15) * 72];
        const unsigned short* bs0 = &Bs[cur][l15 * 72 + klo];
        const unsigned short* bs1 = &Bs[cur][(l15 + 16) * 72 + klo];

        #pragma unroll
        for (int s = 0; s < 2; ++s) {
            FragBF16 a, b0, b1;
            a.q[0]  = *(const uint4*)(asr + s * 32 + kb);        // K = kb..kb+7
            a.q[1]  = *(const uint4*)(asr + s * 32 + kb + 8);    // K = kb+8..kb+15
            b0.q[0] = *(const uint4*)(bs0 + s * 32);             // K = klo..klo+7
            b0.q[1] = *(const uint4*)(bs0 + s * 32 + 8);         // K = klo+8..klo+15
            b1.q[0] = *(const uint4*)(bs1 + s * 32);
            b1.q[1] = *(const uint4*)(bs1 + s * 32 + 8);

            acc0 = __builtin_amdgcn_wmma_f32_16x16x32_bf16(false, a.v, false, b0.v,
                                                           (short)0, acc0, false, false);
            acc1 = __builtin_amdgcn_wmma_f32_16x16x32_bf16(false, a.v, false, b1.v,
                                                           (short)0, acc1, false, false);
        }
        cur ^= 1;
    }

    // --- store C (f32 16x16 layout: VGPR v -> M = lh*8 + v, N = l15)
    const int outRow0 = rowBase + wave * 16 + lh * 8;
    const int col0 = colBase + l15;
    const int col1 = colBase + 16 + l15;
    #pragma unroll
    for (int v = 0; v < 8; ++v) {
        int row = outRow0 + v;
        if (row < nrows) {
            if (col0 < M) {
                float x = acc0[v] + bias[col0];
                if (ACT) x = (x >= 0.0f) ? x : slope * x;
                C[(long)row * ldOut + col0] = x;
            }
            if (col1 < M) {
                float x = acc1[v] + bias[col1];
                if (ACT) x = (x >= 0.0f) ? x : slope * x;
                C[(long)row * ldOut + col1] = x;
            }
        }
    }
}

// ---------------------------------------------------------------------------
// BatchNorm (training-mode batch stats over N rows, HID=256 cols)
// ---------------------------------------------------------------------------
__global__ __launch_bounds__(256) void k_bn_stats(const float* __restrict__ h,
                                                  float* sum, float* sumsq, int nrows) {
    int c = threadIdx.x;                       // column
    long r0 = (long)blockIdx.x * 256;
    long rend = r0 + 256; if (rend > nrows) rend = nrows;
    float s = 0.0f, s2 = 0.0f;
    for (long r = r0; r < rend; ++r) {
        float v = h[r * HID + c];
        s += v; s2 += v * v;
    }
    atomicAdd(&sum[c], s);
    atomicAdd(&sumsq[c], s2);
}

__global__ __launch_bounds__(256) void k_bn_finalize(const float* sum, const float* sumsq,
                                                     const float* gamma, const float* beta,
                                                     float* scale, float* shift, float invN) {
    int c = threadIdx.x;
    float mu  = sum[c] * invN;
    float var = sumsq[c] * invN - mu * mu;
    float sc  = gamma[c] * rsqrtf(var + 1e-5f);
    scale[c] = sc;
    shift[c] = beta[c] - mu * sc;
}

__global__ __launch_bounds__(256) void k_bn_apply_bf16(const float* __restrict__ h,
                                                       const float* __restrict__ scale,
                                                       const float* __restrict__ shift,
                                                       unsigned short* __restrict__ out, long n) {
    long i = (long)blockIdx.x * 256 + threadIdx.x;
    long stride = (long)gridDim.x * 256;
    for (; i < n; i += stride) {
        int c = (int)(i & (HID - 1));
        out[i] = f2bf(h[i] * scale[c] + shift[c]);
    }
}

// leaky/prelu (+ convert to bf16). slope_p!=nullptr -> read slope from device.
__global__ __launch_bounds__(256) void k_leaky_to_bf16(const float* __restrict__ in,
                                                       unsigned short* __restrict__ out, long n,
                                                       const float* slope_p, float slope_c) {
    float s = slope_p ? slope_p[0] : slope_c;
    long i = (long)blockIdx.x * 256 + threadIdx.x;
    long stride = (long)gridDim.x * 256;
    for (; i < n; i += stride) {
        float v = in[i];
        out[i] = f2bf((v >= 0.0f) ? v : s * v);
    }
}

__global__ __launch_bounds__(256) void k_prelu_inplace(float* p, long n, const float* alpha_p) {
    float a = alpha_p[0];
    long i = (long)blockIdx.x * 256 + threadIdx.x;
    long stride = (long)gridDim.x * 256;
    for (; i < n; i += stride) {
        float v = p[i];
        p[i] = (v >= 0.0f) ? v : a * v;
    }
}

// ---------------------------------------------------------------------------
// SpMM scatter: out[rows[e], :] += vals[e] * Min[(perm?)(cols[e]), :]
// one block per edge
// ---------------------------------------------------------------------------
__global__ __launch_bounds__(256) void k_spmm(const int* __restrict__ rows,
                                              const int* __restrict__ cols,
                                              const float* __restrict__ vals,
                                              const float* __restrict__ Min,
                                              float* __restrict__ out,
                                              int C, const int* __restrict__ perm) {
    int e = blockIdx.x;
    int dst = rows[e];
    int src = cols[e];
    if (perm) src = perm[src];
    float v = vals[e];
    const float* mrow = Min + (long)src * C;
    float* orow = out + (long)dst * C;
    for (int c = threadIdx.x; c < C; c += 256)
        atomicAdd(&orow[c], v * mrow[c]);
}

// agg: agg[agg_rows[i], :] += agg_vals[i] * xp[i, :]   (one block per node)
__global__ __launch_bounds__(256) void k_agg(const int* __restrict__ arows,
                                             const float* __restrict__ avals,
                                             const float* __restrict__ xp,
                                             float* __restrict__ agg) {
    int i = blockIdx.x;
    int r = arows[i];
    float v = avals[i];
    const float* xr = xp + (long)i * OUT;
    float* ar = agg + (long)r * OUT;
    for (int c = threadIdx.x; c < OUT; c += 256)
        atomicAdd(&ar[c], v * xr[c]);
}

__global__ __launch_bounds__(256) void k_mse(const float* __restrict__ a,
                                             const float* __restrict__ y,
                                             float* acc, long n) {
    __shared__ float red[256];
    float s = 0.0f;
    long i = (long)blockIdx.x * 256 + threadIdx.x;
    long stride = (long)gridDim.x * 256;
    for (; i < n; i += stride) {
        float d = a[i] - y[i];
        s += d * d;
    }
    red[threadIdx.x] = s;
    __syncthreads();
    for (int o = 128; o > 0; o >>= 1) {
        if ((int)threadIdx.x < o) red[threadIdx.x] += red[threadIdx.x + o];
        __syncthreads();
    }
    if (threadIdx.x == 0) atomicAdd(acc, red[0]);
}

// column sums over h (nrows x C), C <= 288, block = 288 threads
__global__ void k_colsum(const float* __restrict__ h, float* csum, int nrows, int C) {
    int c = threadIdx.x;
    if (c >= C) return;
    long r0 = (long)blockIdx.x * 256;
    long rend = r0 + 256; if (rend > nrows) rend = nrows;
    float s = 0.0f;
    for (long r = r0; r < rend; ++r) s += h[r * C + c];
    atomicAdd(&csum[c], s);
}

__global__ void k_c_finalize(const float* csum, float* cvec, float* cnorm, float invN, int C) {
    __shared__ float red[288];
    int c = threadIdx.x;
    float v = 0.0f;
    if (c < C) { v = csum[c] * invN; cvec[c] = v; }
    red[c] = v * v;
    __syncthreads();
    if (c == 0) {
        float s = 0.0f;
        for (int i = 0; i < C; ++i) s += red[i];
        cnorm[0] = sqrtf(s);
    }
}

// cosine losses: one wave32 per row. mode==0: sum(1-cos), mode==1: sum(max(cos,0))
__global__ __launch_bounds__(256) void k_cos(const float* __restrict__ h,
                                             const float* __restrict__ cvec,
                                             const float* __restrict__ cnorm,
                                             float* acc, int nrows, int C, int mode) {
    __shared__ float red[8];
    int wave = threadIdx.x >> 5, lane = threadIdx.x & 31;
    int row = blockIdx.x * 8 + wave;
    float contrib = 0.0f;
    if (row < nrows) {
        const float* hr = h + (long)row * C;
        float dot = 0.0f, n2 = 0.0f;
        for (int c = lane; c < C; c += 32) {
            float v = hr[c];
            dot += v * cvec[c];
            n2  += v * v;
        }
        for (int o = 16; o > 0; o >>= 1) {
            dot += __shfl_xor(dot, o, 32);
            n2  += __shfl_xor(n2,  o, 32);
        }
        float cs = dot / fmaxf(sqrtf(n2) * cnorm[0], 1e-8f);
        contrib = mode ? fmaxf(cs, 0.0f) : (1.0f - cs);
    }
    if (lane == 0) red[wave] = contrib;
    __syncthreads();
    if (threadIdx.x == 0) {
        float s = 0.0f;
        for (int i = 0; i < 8; ++i) s += red[i];
        atomicAdd(acc, s);
    }
}

__global__ void k_final(const float* acc, float* out, float invSO, float invN) {
    if (threadIdx.x == 0 && blockIdx.x == 0)
        out[0] = acc[0] * invSO + acc[1] * invN + acc[2] * invN;
}

// ---------------------------------------------------------------------------
// Host: workspace layout + launch sequence
// ---------------------------------------------------------------------------
static inline char* carve(char*& p, size_t bytes) {
    char* r = p;
    size_t b = (bytes + 255) & ~(size_t)255;
    p += b;
    return r;
}

extern "C" void kernel_launch(void* const* d_in, const int* in_sizes, int n_in,
                              void* d_out, int out_size, void* d_ws, size_t ws_size,
                              hipStream_t stream) {
    (void)in_sizes; (void)n_in; (void)out_size; (void)ws_size;

    const float* x        = (const float*)d_in[0];
    const float* y        = (const float*)d_in[1];
    const int*   adj_rows = (const int*)  d_in[2];
    const int*   adj_cols = (const int*)  d_in[3];
    const float* adj_vals = (const float*)d_in[4];
    const int*   agg_rows = (const int*)  d_in[5];
    const float* agg_vals = (const float*)d_in[6];
    const int*   perm     = (const int*)  d_in[7];
    const float* mlp_W    = (const float*)d_in[8];
    const float* mlp_b    = (const float*)d_in[9];
    const float* bn_gamma = (const float*)d_in[10];
    const float* bn_beta  = (const float*)d_in[11];
    const float* hg_W1    = (const float*)d_in[12];
    const float* hg_b1    = (const float*)d_in[13];
    const float* hg_W2    = (const float*)d_in[14];
    const float* hg_b2    = (const float*)d_in[15];
    const float* hg_prelu = (const float*)d_in[16];
    const float* pred_W   = (const float*)d_in[17];
    const float* pred_b   = (const float*)d_in[18];
    const float* dgi_W    = (const float*)d_in[19];
    const float* dgi_b    = (const float*)d_in[20];
    const float* dgi_prelu= (const float*)d_in[21];

    float* loss_out = (float*)d_out;
    float* x_prime  = loss_out + 1;      // N x OUT

    // -------- workspace carving --------
    char* p = (char*)d_ws;
    unsigned short* XBF   = (unsigned short*)carve(p, (size_t)N_NODES * F_IN * 2);
    unsigned short* WMLP  = (unsigned short*)carve(p, (size_t)HID * F_IN * 2);   // 256 x 1024 (T)
    unsigned short* W1    = (unsigned short*)carve(p, (size_t)HID * HID * 2);    // 256 x 256 (T)
    unsigned short* W2    = (unsigned short*)carve(p, (size_t)HID * HID * 2);
    unsigned short* WPRED = (unsigned short*)carve(p, (size_t)OUTP * HID * 2);   // 288 x 256 (T)
    unsigned short* WDGI  = (unsigned short*)carve(p, (size_t)OUTP * HID * 2);
    float*          H0RAW = (float*)carve(p, (size_t)N_NODES * HID * 4);
    unsigned short* H0BF  = (unsigned short*)carve(p, (size_t)N_NODES * HID * 2);
    unsigned short* TBF   = (unsigned short*)carve(p, (size_t)N_NODES * HID * 2);
    float*          MBUF  = (float*)carve(p, (size_t)N_NODES * OUTP * 4);   // m1/m2/m3
    float*          TOUT  = (float*)carve(p, (size_t)N_NODES * HID * 4);    // t / t2
    float*          H1    = (float*)carve(p, (size_t)N_NODES * OUT * 4);
    float*          H2    = (float*)carve(p, (size_t)N_NODES * OUT * 4);
    float*          AGG   = (float*)carve(p, (size_t)S_SP * OUT * 4);
    float*          SMALL = (float*)carve(p, 2048 * 4);
    float* bn_sum   = SMALL;            // 256
    float* bn_sumsq = SMALL + 256;      // 256
    float* bn_scale = SMALL + 512;      // 256
    float* bn_shift = SMALL + 768;      // 256
    float* csum     = SMALL + 1024;     // 288
    float* cvec     = SMALL + 1312;     // 288
    float* cnorm    = SMALL + 1600;     // 1
    float* acc      = SMALL + 1604;     // 3: mse_sum, cos1_sum, cos2_sum

    const float invN  = 1.0f / (float)N_NODES;
    const float invSO = 1.0f / (float)(S_SP * OUT);

    // 1) zero small accumulators
    k_zero<<<8, 256, 0, stream>>>(SMALL, 2048);

    // 2) bf16 conversions / transposed weight padding
    k_f32_to_bf16<<<4096, 256, 0, stream>>>(x, XBF, (long)N_NODES * F_IN);
    k_pad_weight_t<<<1024, 256, 0, stream>>>(mlp_W, WMLP, F_IN, HID, HID);
    k_pad_weight_t<<<256, 256, 0, stream>>>(hg_W1, W1, HID, HID, HID);
    k_pad_weight_t<<<256, 256, 0, stream>>>(hg_W2, W2, HID, HID, HID);
    k_pad_weight_t<<<288, 256, 0, stream>>>(pred_W, WPRED, HID, OUT, OUTP);
    k_pad_weight_t<<<288, 256, 0, stream>>>(dgi_W, WDGI, HID, OUT, OUTP);

    const dim3 gH(HID / 32,  (N_NODES + 127) / 128);   // 8 x 391
    const dim3 gO(OUTP / 32, (N_NODES + 127) / 128);   // 9 x 391

    // 3) h0_raw = leaky(x @ mlp_W + b, 0.1)
    k_gemm_bf16<1><<<gH, 256, 0, stream>>>(XBF, WMLP, mlp_b, H0RAW,
                                           N_NODES, F_IN, HID, HID, 0.1f);
    // 4) batchnorm -> h0 (bf16)
    k_bn_stats<<<(N_NODES + 255) / 256, 256, 0, stream>>>(H0RAW, bn_sum, bn_sumsq, N_NODES);
    k_bn_finalize<<<1, 256, 0, stream>>>(bn_sum, bn_sumsq, bn_gamma, bn_beta,
                                         bn_scale, bn_shift, invN);
    k_bn_apply_bf16<<<4096, 256, 0, stream>>>(H0RAW, bn_scale, bn_shift, H0BF,
                                              (long)N_NODES * HID);

    // 5) HGNN layer 1: m1 = h0 @ W1 + b1 ; t = spmm(m1) ; prelu
    k_gemm_bf16<0><<<gH, 256, 0, stream>>>(H0BF, W1, hg_b1, MBUF,
                                           N_NODES, HID, HID, HID, 0.0f);
    k_zero<<<4096, 256, 0, stream>>>(TOUT, (long)N_NODES * HID);
    k_spmm<<<NNZ, 256, 0, stream>>>(adj_rows, adj_cols, adj_vals, MBUF, TOUT, HID, nullptr);
    k_leaky_to_bf16<<<4096, 256, 0, stream>>>(TOUT, TBF, (long)N_NODES * HID,
                                              hg_prelu, 0.0f);

    // 6) HGNN layer 2: m2 = t @ W2 + b2 ; t2 = spmm(m2) ; h = leaky(t2, 0.01)
    k_gemm_bf16<0><<<gH, 256, 0, stream>>>(TBF, W2, hg_b2, MBUF,
                                           N_NODES, HID, HID, HID, 0.0f);
    k_zero<<<4096, 256, 0, stream>>>(TOUT, (long)N_NODES * HID);
    k_spmm<<<NNZ, 256, 0, stream>>>(adj_rows, adj_cols, adj_vals, MBUF, TOUT, HID, nullptr);
    k_leaky_to_bf16<<<4096, 256, 0, stream>>>(TOUT, TBF, (long)N_NODES * HID,
                                              nullptr, 0.01f);

    // 7) x_prime = leaky(h @ pred_W + pred_b, 0.01) -> d_out+1
    k_gemm_bf16<1><<<gO, 256, 0, stream>>>(TBF, WPRED, pred_b, x_prime,
                                           N_NODES, HID, OUT, OUT, 0.01f);

    // 8) DGI: m3 = h0 @ dgi_W + b ; h1 = prelu(spmm(m3)) ; h2 = prelu(spmm(perm-gather m3))
    //    (row permutation commutes with the right-multiply+bias: m3p[i] = m3[perm[i]])
    k_gemm_bf16<0><<<gO, 256, 0, stream>>>(H0BF, WDGI, dgi_b, MBUF,
                                           N_NODES, HID, OUT, OUT, 0.0f);
    k_zero<<<4096, 256, 0, stream>>>(H1, (long)N_NODES * OUT);
    k_zero<<<4096, 256, 0, stream>>>(H2, (long)N_NODES * OUT);
    k_spmm<<<NNZ, 256, 0, stream>>>(adj_rows, adj_cols, adj_vals, MBUF, H1, OUT, nullptr);
    k_spmm<<<NNZ, 256, 0, stream>>>(adj_rows, adj_cols, adj_vals, MBUF, H2, OUT, perm);
    k_prelu_inplace<<<4096, 256, 0, stream>>>(H1, (long)N_NODES * OUT, dgi_prelu);
    k_prelu_inplace<<<4096, 256, 0, stream>>>(H2, (long)N_NODES * OUT, dgi_prelu);

    // 9) c = mean(h1, axis=0); c_norm
    k_colsum<<<(N_NODES + 255) / 256, 288, 0, stream>>>(H1, csum, N_NODES, OUT);
    k_c_finalize<<<1, 288, 0, stream>>>(csum, cvec, cnorm, invN, OUT);

    // 10) cosine losses
    k_cos<<<(N_NODES + 7) / 8, 256, 0, stream>>>(H1, cvec, cnorm, acc + 1,
                                                 N_NODES, OUT, 0);
    k_cos<<<(N_NODES + 7) / 8, 256, 0, stream>>>(H2, cvec, cnorm, acc + 2,
                                                 N_NODES, OUT, 1);

    // 11) Xenium MSE: agg = segsum(agg_vals * x_prime) ; mse
    k_zero<<<2048, 256, 0, stream>>>(AGG, (long)S_SP * OUT);
    k_agg<<<N_NODES, 256, 0, stream>>>(agg_rows, agg_vals, x_prime, AGG);
    k_mse<<<1024, 256, 0, stream>>>(AGG, y, acc, (long)S_SP * OUT);

    // 12) final loss
    k_final<<<1, 32, 0, stream>>>(acc, loss_out, invSO, invN);
}
